// DynamicQueryAttention_61134564491826
// MI455X (gfx1250) — matrix operand order repaired and checked by reading, not compiled, not run on hardware
//
#include <hip/hip_runtime.h>
#include <hip/hip_bf16.h>
#include <math.h>

// ---------------------------------------------------------------------------
// DynamicQueryAttention, fully fused for MI455X (gfx1250, wave32, WMMA).
//
// Math factorization (removes the serial dependency through `summed`):
//   k_v = v_dom @ Wk_v^T + bk_v          [B,64]
//   k_t = t_dom @ Wk_t^T + bk_t          [B,64]
//   q   = gw    @ Wq^T   + bq            [B,64]
//   e_v = v_enc @ Wq^T                   [B,64]
//   e_t = t_enc @ Wq^T                   [B,64]
//   s   = softmax([k_v.q, k_t.q])
//   q2  = s0*e_v + s1*e_t + bq
//   out = softmax([k_v.q2, k_t.q2])
//
// Five [B,512]x[512,64] GEMMs, one pass over all inputs -> HBM-roofline bound.
// f32 inputs converted to bf16 in registers, v_wmma_f32_16x16x32_bf16, f32 acc.
// ---------------------------------------------------------------------------

typedef __attribute__((ext_vector_type(16))) __bf16 v16bf;
typedef __attribute__((ext_vector_type(8)))  float  v8f;

#define DQA_D       512
#define DQA_H       64
#define DQA_KSTEPS  16      // D / 32
#define DQA_NT      4       // H / 16
#define WAVES_PB    2       // waves per block (64 threads)
#define ROWS_PW     16      // rows per wave
#define LDS_STRIDE  66      // 8*66 % 64 == 16 -> the two lane-half row groups
                            // land on disjoint LDS banks

// --- Kernel 0: convert the 3 weight matrices (H x D, f32, row-major) into a
// bf16 buffer pre-swizzled into WMMA B-fragment order.
// Layout: [w][kstep][ntile][lane][j], j = K within the lane's 16 values.
//   n = ntile*16 + (lane & 15)
//   k = kstep*32 + ((lane >> 4) & 1)*16 + j
__global__ void dqa_wconvert(const float* __restrict__ Wq,
                             const float* __restrict__ Wk_v,
                             const float* __restrict__ Wk_t,
                             __bf16* __restrict__ out) {
    int tid = blockIdx.x * blockDim.x + threadIdx.x;
    if (tid >= 3 * DQA_H * DQA_D) return;
    int j     =  tid        & 15;
    int lane  = (tid >> 4)  & 31;
    int nt    = (tid >> 9)  & 3;
    int kstep = (tid >> 11) & 15;
    int w     =  tid >> 15;
    int n = nt * 16 + (lane & 15);
    int k = kstep * 32 + ((lane >> 4) & 1) * 16 + j;
    const float* W = (w == 0) ? Wq : (w == 1) ? Wk_v : Wk_t;
    out[tid] = (__bf16)W[n * DQA_D + k];
}

// --- Main fused kernel: 64 threads = 2 waves, each wave owns 16 rows of B.
__global__ __launch_bounds__(64, 1)
void dqa_main(const float* __restrict__ v_dom, const float* __restrict__ t_dom,
              const float* __restrict__ v_enc, const float* __restrict__ t_enc,
              const float* __restrict__ gw,
              const float* __restrict__ bq,
              const float* __restrict__ bk_v, const float* __restrict__ bk_t,
              const __bf16* __restrict__ wbuf,
              float* __restrict__ out) {
    __shared__ float smem[WAVES_PB][5][ROWS_PW][LDS_STRIDE];

    const int wave = threadIdx.x >> 5;
    const int lane = threadIdx.x & 31;
    const int m    = lane & 15;          // row within tile (A layout), column (C layout)
    const int hi   = lane >> 4;          // 0: lanes 0-15, 1: lanes 16-31
    const long rowBase = (long)(blockIdx.x * WAVES_PB + wave) * ROWS_PW;

    // 5 GEMM pairs: (input, weight-slot in wbuf, bias)
    #pragma unroll 1
    for (int p = 0; p < 5; ++p) {
        const float* inp  = (p == 0) ? v_dom : (p == 1) ? t_dom
                          : (p == 2) ? gw    : (p == 3) ? v_enc : t_enc;
        const float* bias = (p == 0) ? bk_v  : (p == 1) ? bk_t
                          : (p == 2) ? bq    : nullptr;
        const int    wsel = (p == 0) ? 1 : (p == 1) ? 2 : 0;

        // A: this lane streams row (rowBase+m); lanes 16-31 carry K offset +16.
        const float*  rp = inp + (rowBase + m) * DQA_D + hi * 16;
        const v16bf*  wb = (const v16bf*)(wbuf + (size_t)wsel * (16 * 4 * 32 * 16));

        v8f acc0 = {}, acc1 = {}, acc2 = {}, acc3 = {};

        #pragma unroll 2
        for (int ks = 0; ks < DQA_KSTEPS; ++ks) {
            const float* a0 = rp + ks * 32;
            __builtin_prefetch(a0 + 64, 0, 0);   // global_prefetch_b8, 2 steps ahead

            float fa[16];
            *(float4*)(fa + 0)  = *(const float4*)(a0 + 0);
            *(float4*)(fa + 4)  = *(const float4*)(a0 + 4);
            *(float4*)(fa + 8)  = *(const float4*)(a0 + 8);
            *(float4*)(fa + 12) = *(const float4*)(a0 + 12);

            v16bf a;
            #pragma unroll
            for (int j = 0; j < 16; ++j) a[j] = (__bf16)fa[j];

            const v16bf* wrow = wb + (size_t)(ks * DQA_NT) * 32 + lane;
            v16bf b0 = wrow[0 * 32];
            v16bf b1 = wrow[1 * 32];
            v16bf b2 = wrow[2 * 32];
            v16bf b3 = wrow[3 * 32];

            acc0 = __builtin_amdgcn_wmma_f32_16x16x32_bf16(false, a, false, b0,
                                                           (short)0, acc0, false, false);
            acc1 = __builtin_amdgcn_wmma_f32_16x16x32_bf16(false, a, false, b1,
                                                           (short)0, acc1, false, false);
            acc2 = __builtin_amdgcn_wmma_f32_16x16x32_bf16(false, a, false, b2,
                                                           (short)0, acc2, false, false);
            acc3 = __builtin_amdgcn_wmma_f32_16x16x32_bf16(false, a, false, b3,
                                                           (short)0, acc3, false, false);
        }

        // Bias (per output column) + spill C tiles to LDS.
        // C layout: VGPR i -> row (i + 8*hi), col = nt*16 + m.
        const int crow = 8 * hi;
        float bv0 = bias ? bias[0 * 16 + m] : 0.0f;
        float bv1 = bias ? bias[1 * 16 + m] : 0.0f;
        float bv2 = bias ? bias[2 * 16 + m] : 0.0f;
        float bv3 = bias ? bias[3 * 16 + m] : 0.0f;
        #pragma unroll
        for (int i = 0; i < 8; ++i) {
            smem[wave][p][crow + i][0 * 16 + m] = acc0[i] + bv0;
            smem[wave][p][crow + i][1 * 16 + m] = acc1[i] + bv1;
            smem[wave][p][crow + i][2 * 16 + m] = acc2[i] + bv2;
            smem[wave][p][crow + i][3 * 16 + m] = acc3[i] + bv3;
        }
    }

    __syncthreads();

    // --- Per-row epilogue: lanes 0-15 each own one row of the wave tile.
    if (lane < 16) {
        const long  row = rowBase + lane;
        const float* kv = &smem[wave][0][lane][0];
        const float* kt = &smem[wave][1][lane][0];
        const float* qq = &smem[wave][2][lane][0];
        const float* ev = &smem[wave][3][lane][0];
        const float* et = &smem[wave][4][lane][0];

        float dv = 0.0f, dt = 0.0f;
        #pragma unroll 8
        for (int h = 0; h < DQA_H; ++h) {
            dv += kv[h] * qq[h];
            dt += kt[h] * qq[h];
        }
        float mx = fmaxf(dv, dt);
        float e0 = __expf(dv - mx), e1 = __expf(dt - mx);
        float s0 = e0 / (e0 + e1), s1 = 1.0f - s0;

        float dv2 = 0.0f, dt2 = 0.0f;
        #pragma unroll 8
        for (int h = 0; h < DQA_H; ++h) {
            float q2 = s0 * ev[h] + s1 * et[h] + bq[h];
            dv2 += kv[h] * q2;
            dt2 += kt[h] * q2;
        }
        mx = fmaxf(dv2, dt2);
        e0 = __expf(dv2 - mx); e1 = __expf(dt2 - mx);
        float inv = 1.0f / (e0 + e1);
        *(float2*)&out[row * 2] = make_float2(e0 * inv, e1 * inv);
    }
}

extern "C" void kernel_launch(void* const* d_in, const int* in_sizes, int n_in,
                              void* d_out, int out_size, void* d_ws, size_t ws_size,
                              hipStream_t stream) {
    // setup_inputs order:
    // 0 v_dom, 1 t_dom, 2 v_enc, 3 t_enc, 4 Wq, 5 bq, 6 Wk_v, 7 bk_v,
    // 8 Wk_t, 9 bk_t, 10 gw_state
    const float* v_dom = (const float*)d_in[0];
    const float* t_dom = (const float*)d_in[1];
    const float* v_enc = (const float*)d_in[2];
    const float* t_enc = (const float*)d_in[3];
    const float* Wq    = (const float*)d_in[4];
    const float* bq    = (const float*)d_in[5];
    const float* Wk_v  = (const float*)d_in[6];
    const float* bk_v  = (const float*)d_in[7];
    const float* Wk_t  = (const float*)d_in[8];
    const float* bk_t  = (const float*)d_in[9];
    const float* gw    = (const float*)d_in[10];
    float* out = (float*)d_out;

    __bf16* wbuf = (__bf16*)d_ws;   // needs 3*64*512*2 = 192 KB of workspace
    (void)ws_size; (void)n_in; (void)out_size;

    const int B = in_sizes[0] / DQA_D;

    // Stage weights as bf16 in WMMA B-fragment order.
    {
        int n = 3 * DQA_H * DQA_D;   // 98304
        dqa_wconvert<<<(n + 255) / 256, 256, 0, stream>>>(Wq, Wk_v, Wk_t, wbuf);
    }

    // Fused GEMMs + softmax epilogue.
    {
        int blocks = B / (WAVES_PB * ROWS_PW);   // 131072 / 32 = 4096
        dqa_main<<<blocks, WAVES_PB * 32, 0, stream>>>(
            v_dom, t_dom, v_enc, t_enc, gw, bq, bk_v, bk_t, wbuf, out);
    }
}
